// CINLayer_47528108098262
// MI455X (gfx1250) — matrix operand order repaired and checked by reading, not compiled, not run on hardware
//
#include <hip/hip_runtime.h>

typedef _Float16 v16h __attribute__((ext_vector_type(16)));
typedef _Float16 v8h  __attribute__((ext_vector_type(8)));
typedef _Float16 v4h  __attribute__((ext_vector_type(4)));
typedef float    v8f  __attribute__((ext_vector_type(8)));
typedef float    v4f  __attribute__((ext_vector_type(4)));

// Problem constants (fixed by the reference)
#define BB 2048
#define CC 64
#define NN 64
#define DD 64
#define FF 128
#define KTOT (CC*NN)          // 4096
#define X0T_PITCH 72          // 64 + pad, keeps 16B alignment (144B rows)
#define GB 4                  // batches per workgroup (amortizes W L2 stream 4x)

// ---------------- W f32 -> f16 conversion (runs once per launch) -------------
__global__ __launch_bounds__(256) void wconvert_kernel(const float* __restrict__ W,
                                                       _Float16* __restrict__ Wh) {
    const int i = (blockIdx.x * 256 + threadIdx.x) * 4;   // FF*CC*NN = 524288, grid=512
    v4f v = *(const v4f*)(W + i);
    v4h h = { (_Float16)v[0], (_Float16)v[1], (_Float16)v[2], (_Float16)v[3] };
    *(v4h*)(Wh + i) = h;
}

static __device__ inline float hsum8(v8f t) {
    return ((t[0] + t[1]) + (t[2] + t[3])) + ((t[4] + t[5]) + (t[6] + t[7]));
}

// ---------- Main fused CIN kernel: one workgroup per GB batches --------------
__global__ __launch_bounds__(256, 1) void cin_kernel(const float* __restrict__ xj,
                                                     const float* __restrict__ x0,
                                                     const _Float16* __restrict__ Wh,
                                                     float* __restrict__ out,
                                                     float* __restrict__ outp) {
    __shared__ _Float16 x0T[GB][DD * X0T_PITCH];  // x0[b] transposed: [d][n], f16
    __shared__ _Float16 xjh[GB][CC * DD];         // xj[b] row-major:  [c][d], f16
    __shared__ float    psum[GB][2][FF];

    const int b0  = blockIdx.x * GB;
    const int tid = threadIdx.x;

    // ---- Stage GB batches into LDS (convert to f16; transpose x0) ----
#pragma unroll
    for (int g = 0; g < GB; ++g) {
        const float* __restrict__ x0b = x0 + (size_t)(b0 + g) * (NN * DD);
        const float* __restrict__ xjb = xj + (size_t)(b0 + g) * (CC * DD);
        const int n = tid >> 2;        // 0..63
        const int q = tid & 3;         // d block of 16
        const float* src = x0b + n * DD + q * 16;
#pragma unroll
        for (int i = 0; i < 4; ++i) {
            v4f v = *(const v4f*)(src + i * 4);
            const int dd = q * 16 + i * 4;
            x0T[g][(dd + 0) * X0T_PITCH + n] = (_Float16)v[0];
            x0T[g][(dd + 1) * X0T_PITCH + n] = (_Float16)v[1];
            x0T[g][(dd + 2) * X0T_PITCH + n] = (_Float16)v[2];
            x0T[g][(dd + 3) * X0T_PITCH + n] = (_Float16)v[3];
        }
        const float* s2 = xjb + tid * 16;
#pragma unroll
        for (int i = 0; i < 4; ++i) {
            v4f v = *(const v4f*)(s2 + i * 4);
            v4h h = { (_Float16)v[0], (_Float16)v[1], (_Float16)v[2], (_Float16)v[3] };
            *(v4h*)(&xjh[g][tid * 16 + i * 4]) = h;
        }
    }
    __syncthreads();

    // ---- Wave tiling: 8 waves = 4 f-groups x 2 d-groups; each wave does a
    //      2x2 grid of 16x16 tiles (M=d, N=f) for each of GB batches. ----
    const int lane = tid & 31;
    const int wave = tid >> 5;
    const int l15  = lane & 15;
    const int hi   = lane >> 4;               // 0: lanes 0-15, 1: lanes 16-31
    const int fb   = (wave & 3) * 32;         // f base (covers fb..fb+31)
    const int mb   = (wave >> 2) * 32;        // d base (covers mb..mb+31)
    const int dg   = wave >> 2;

    v8f acc[GB][2][2];                        // [g][d-tile][f-tile] : 128 VGPRs
#pragma unroll
    for (int g = 0; g < GB; ++g)
#pragma unroll
        for (int a = 0; a < 2; ++a)
#pragma unroll
            for (int f = 0; f < 2; ++f)
                acc[g][a][f] = (v8f){0,0,0,0,0,0,0,0};

    const int dda = mb + l15;                 // d row for d-tile 0 (this lane)
    const int ddb = mb + 16 + l15;            // d row for d-tile 1
    // B operand: per lane 16 contiguous halfs of W[f, c, n_base + hi*16 ..]
    const _Float16* w0 = Wh + (fb      + l15) * KTOT + hi * 16;
    const _Float16* w1 = Wh + (fb + 16 + l15) * KTOT + hi * 16;

    for (int c = 0; c < CC; ++c) {
        _Float16 xva[GB], xvb[GB];
#pragma unroll
        for (int g = 0; g < GB; ++g) {
            xva[g] = xjh[g][c * DD + dda];
            xvb[g] = xjh[g][c * DD + ddb];
        }
        const int cofs = c * NN;
#pragma unroll
        for (int nh = 0; nh < 2; ++nh) {
            const int nb = nh * 32;           // K-step of 32 within fixed c
            v16h bm0 = *(const v16h*)(w0 + cofs + nb);   // 2x global_load_b128 (L2)
            v16h bm1 = *(const v16h*)(w1 + cofs + nb);
#pragma unroll
            for (int g = 0; g < GB; ++g) {
                // A operand: x0 column chunks (ds_load_b128 x2) * xj scalar
                union { v16h v; v8h h[2]; } a0u, a1u;
                const _Float16* xr0 = &x0T[g][dda * X0T_PITCH];
                const _Float16* xr1 = &x0T[g][ddb * X0T_PITCH];
                a0u.h[0] = *(const v8h*)(xr0 + nb + hi * 16);
                a0u.h[1] = *(const v8h*)(xr0 + nb + hi * 16 + 8);
                a1u.h[0] = *(const v8h*)(xr1 + nb + hi * 16);
                a1u.h[1] = *(const v8h*)(xr1 + nb + hi * 16 + 8);
                v16h am0 = a0u.v * xva[g];    // 8x v_pk_mul_f16 (co-exec w/ WMMA)
                v16h am1 = a1u.v * xvb[g];
                acc[g][0][0] = __builtin_amdgcn_wmma_f32_16x16x32_f16(
                    false, am0, false, bm0, (short)0, acc[g][0][0], false, false);
                acc[g][0][1] = __builtin_amdgcn_wmma_f32_16x16x32_f16(
                    false, am0, false, bm1, (short)0, acc[g][0][1], false, false);
                acc[g][1][0] = __builtin_amdgcn_wmma_f32_16x16x32_f16(
                    false, am1, false, bm0, (short)0, acc[g][1][0], false, false);
                acc[g][1][1] = __builtin_amdgcn_wmma_f32_16x16x32_f16(
                    false, am1, false, bm1, (short)0, acc[g][1][1], false, false);
            }
        }
    }

    // ---- Store cin_out (write-once stream -> non-temporal b128 stores) ----
    // C/D layout: lane: N(col f)=l15; vgpr r: M(row d)= r + hi*8 -> 8 consecutive d.
#pragma unroll
    for (int g = 0; g < GB; ++g) {
        float* __restrict__ ob = out + (size_t)(b0 + g) * (FF * DD);
#pragma unroll
        for (int a = 0; a < 2; ++a)
#pragma unroll
            for (int f = 0; f < 2; ++f) {
                float* p = ob + (fb + f * 16 + l15) * DD + (mb + a * 16) + hi * 8;
                v8f t = acc[g][a][f];
                v4f lo = { t[0], t[1], t[2], t[3] };
                v4f hv = { t[4], t[5], t[6], t[7] };
                __builtin_nontemporal_store(lo, (v4f*)p);
                __builtin_nontemporal_store(hv, (v4f*)(p + 4));
            }
    }

    // ---- cin_p_out: reduce over d (vgprs -> lane halves -> cross-wave) ----
#pragma unroll
    for (int g = 0; g < GB; ++g) {
        float sf0 = hsum8(acc[g][0][0]) + hsum8(acc[g][1][0]); // f-tile fb
        float sf1 = hsum8(acc[g][0][1]) + hsum8(acc[g][1][1]); // f-tile fb+16
        sf0 += __shfl_xor(sf0, 16);
        sf1 += __shfl_xor(sf1, 16);
        if (hi == 0) {
            psum[g][dg][fb + l15]      = sf0;
            psum[g][dg][fb + 16 + l15] = sf1;
        }
    }
    __syncthreads();
    if (tid < FF) {
#pragma unroll
        for (int g = 0; g < GB; ++g)
            outp[(size_t)(b0 + g) * FF + tid] = psum[g][0][tid] + psum[g][1][tid];
    }
}

extern "C" void kernel_launch(void* const* d_in, const int* in_sizes, int n_in,
                              void* d_out, int out_size, void* d_ws, size_t ws_size,
                              hipStream_t stream) {
    (void)in_sizes; (void)n_in; (void)out_size; (void)ws_size;
    const float* xj = (const float*)d_in[0];   // (B, C, D) f32
    const float* x0 = (const float*)d_in[1];   // (B, N, D) f32
    const float* W  = (const float*)d_in[2];   // (F, C, N) f32
    float* out  = (float*)d_out;               // cin_out (B,F,D) then cin_p_out (B,F)
    float* outp = out + (size_t)BB * FF * DD;
    _Float16* Wh = (_Float16*)d_ws;            // 1 MB f16 copy of W (L2-resident)

    wconvert_kernel<<<(FF * CC * NN) / (256 * 4), 256, 0, stream>>>(W, Wh);
    cin_kernel<<<BB / GB, 256, 0, stream>>>(xj, x0, Wh, out, outp);
}
